// MoeTokenLevelFeedForwardGshard_30494267801780
// MI455X (gfx1250) — matrix operand order repaired and checked by reading, not compiled, not run on hardware
//
#include <hip/hip_runtime.h>
#include <hip/hip_bf16.h>

// ---------------------------------------------------------------------------
// MoE GShard token-level FFN for MI455X (gfx1250, wave32, WMMA)
//   D_MODEL=512, D_FF=2048, E=4, S=4096, CAP=S/E=1024
// Sparse capacity dispatch + bf16 WMMA GEMMs with fp32 accumulation.
// GEMMs use a 64x64 wave tile: 16 v_wmma per K-step @ 16 b128 loads.
// ---------------------------------------------------------------------------

#define S_TOK 4096
#define DM    512
#define DFF   2048
#define NE    4
#define CAP   1024
#define Y_ELEMS (S_TOK * DM)
#define EPS_F 2.2204460492503131e-16f

typedef __attribute__((ext_vector_type(16))) __bf16 v16bf;
typedef __attribute__((ext_vector_type(8)))  __bf16 v8bf;
typedef __attribute__((ext_vector_type(8)))  float  v8f;

union FragU { v16bf v; v8bf h[2]; };

__device__ __forceinline__ unsigned short f2bf(float f) {
  union { float f; unsigned u; } v; v.f = f;
  unsigned r = v.u + 0x7FFFu + ((v.u >> 16) & 1u);   // round-to-nearest-even
  return (unsigned short)(r >> 16);
}

// A-fragment (16x32 bf16): lanes 0-15 hold row M=lane, K=[0..8)+[16..24);
// lanes 16-31 hold row M=lane-16, K=[8..16)+[24..32)  -> two 16B chunks.
__device__ __forceinline__ v16bf load_a(const unsigned short* base, int row,
                                        int ld, int k, int lane) {
  const unsigned short* p =
      base + (size_t)(row + (lane & 15)) * ld + k + ((lane >> 4) << 3);
  FragU f;
  f.h[0] = *reinterpret_cast<const v8bf*>(p);
  f.h[1] = *reinterpret_cast<const v8bf*>(p + 16);
  return f.v;
}

// B-fragment (32x16 bf16), B stored transposed [N][K]: lanes 0-15 hold
// column N=lane with K=[0..8)+[16..24); lanes 16-31 K=[8..16)+[24..32) -> two 16B chunks.
__device__ __forceinline__ v16bf load_b(const unsigned short* baseT, int col,
                                        int ld, int k, int lane) {
  const unsigned short* p =
      baseT + (size_t)(col + (lane & 15)) * ld + k + ((lane >> 4) << 3);
  FragU f;
  f.h[0] = *reinterpret_cast<const v8bf*>(p);
  f.h[1] = *reinterpret_cast<const v8bf*>(p + 16);
  return f.v;
}

__device__ __forceinline__ v8f wmma_bf(v16bf a, v16bf b, v8f c) {
  return __builtin_amdgcn_wmma_f32_16x16x32_bf16(
      /*neg_a=*/false, a, /*neg_b=*/false, b,
      /*c_mod=*/(short)0, c, /*reuse_a=*/false, /*reuse_b=*/false);
}

// ---------------------------------------------------------------------------
// 1) init: zero y, token index table, gate value table
// ---------------------------------------------------------------------------
__global__ __launch_bounds__(256) void moe_init(float* __restrict__ y,
                                                int* __restrict__ tok,
                                                float* __restrict__ gv) {
  const int idx = blockIdx.x * 256 + threadIdx.x;
  if (idx < Y_ELEMS) y[idx] = 0.0f;
  if (idx < NE * CAP) { tok[idx] = 0; gv[idx] = 0.0f; }
}

// ---------------------------------------------------------------------------
// 2) gating: one wave per token. logits = x[s,:] @ w_gate; softmax; top-2.
// ---------------------------------------------------------------------------
__global__ __launch_bounds__(128) void moe_gate(const float* __restrict__ x,
                                                const float* __restrict__ rand_u,
                                                const float* __restrict__ w_gate,
                                                int* __restrict__ e1a,
                                                int* __restrict__ e2a,
                                                float* __restrict__ g1a,
                                                float* __restrict__ g2a,
                                                float* __restrict__ probs) {
  const int lane = threadIdx.x & 31;
  const int s = blockIdx.x * 4 + (threadIdx.x >> 5);
  const float4* xs = reinterpret_cast<const float4*>(x + (size_t)s * DM);
  float l[4] = {0.f, 0.f, 0.f, 0.f};
#pragma unroll
  for (int j = 0; j < 4; ++j) {
    float4 xv = xs[lane * 4 + j];
    const float* wg = w_gate + (lane * 16 + j * 4) * NE;
#pragma unroll
    for (int e = 0; e < NE; ++e) {
      l[e] += xv.x * wg[0 + e] + xv.y * wg[4 + e] +
              xv.z * wg[8 + e] + xv.w * wg[12 + e];
    }
  }
#pragma unroll
  for (int off = 16; off > 0; off >>= 1) {
#pragma unroll
    for (int e = 0; e < NE; ++e) l[e] += __shfl_xor(l[e], off, 32);
  }
  if (lane == 0) {
    float m = fmaxf(fmaxf(l[0], l[1]), fmaxf(l[2], l[3]));
    float p[4], sum = 0.f;
#pragma unroll
    for (int e = 0; e < NE; ++e) { p[e] = __expf(l[e] - m); sum += p[e]; }
#pragma unroll
    for (int e = 0; e < NE; ++e) { p[e] /= sum; probs[s * NE + e] = p[e]; }
    int i1 = 0;
    for (int e = 1; e < NE; ++e) if (p[e] > p[i1]) i1 = e;
    int i2 = (i1 == 0) ? 1 : 0;
    for (int e = 0; e < NE; ++e) if (e != i1 && p[e] > p[i2]) i2 = e;
    const float denom = p[i1] + p[i2];
    const float g1 = p[i1] / denom, g2 = p[i2] / denom;
    const bool pass2 = g2 > rand_u[s * NE + i2] * 0.5f;
    e1a[s] = i1;                 g1a[s] = g1;
    e2a[s] = pass2 ? i2 : -1;    g2a[s] = g2;
  }
}

// ---------------------------------------------------------------------------
// 3) deterministic capacity assignment + loss statistics (one thread/expert)
// ---------------------------------------------------------------------------
__global__ __launch_bounds__(32) void moe_assign(const int* __restrict__ e1a,
                                                 const int* __restrict__ e2a,
                                                 const float* __restrict__ g1a,
                                                 const float* __restrict__ g2a,
                                                 const float* __restrict__ probs,
                                                 int* __restrict__ tok,
                                                 float* __restrict__ gv,
                                                 int* __restrict__ nfill,
                                                 float* __restrict__ cnt1,
                                                 float* __restrict__ psum) {
  const int e = threadIdx.x;
  if (e >= NE) return;
  int cnt = 0;
  float c1 = 0.f, ps = 0.f;
  for (int s = 0; s < S_TOK; ++s) {
    ps += probs[s * NE + e];
    if (e1a[s] == e) {
      c1 += 1.0f;
      if (cnt < CAP) { tok[e * CAP + cnt] = s; gv[e * CAP + cnt] = g1a[s]; }
      ++cnt;
    }
    if (e2a[s] == e) {
      if (cnt < CAP) { tok[e * CAP + cnt] = s; gv[e * CAP + cnt] = g2a[s]; }
      ++cnt;
    }
  }
  nfill[e] = (cnt < CAP) ? cnt : CAP;
  cnt1[e] = c1;
  psum[e] = ps;
}

// ---------------------------------------------------------------------------
// 4) weight convert: fp32 -> bf16, transposed to [E][N][K] for B fragments
// ---------------------------------------------------------------------------
__global__ __launch_bounds__(256) void moe_cvt_w(const float* __restrict__ W1,
                                                 const float* __restrict__ W2,
                                                 unsigned short* __restrict__ W1bT,
                                                 unsigned short* __restrict__ W2bT) {
  const int idx = blockIdx.x * 256 + threadIdx.x;  // < NE*DM*DFF
  {  // W1bT[e][n<DFF][k<DM] = W1[e][k][n]
    const int k = idx & (DM - 1);
    const int n = (idx / DM) & (DFF - 1);
    const int e = idx / (DM * DFF);
    W1bT[idx] = f2bf(W1[((size_t)e * DM + k) * DFF + n]);
  }
  {  // W2bT[e][n<DM][k<DFF] = W2[e][k][n]
    const int k = idx & (DFF - 1);
    const int n = (idx / DFF) & (DM - 1);
    const int e = idx / (DM * DFF);
    W2bT[idx] = f2bf(W2[((size_t)e * DFF + k) * DM + n]);
  }
}

// ---------------------------------------------------------------------------
// 5) dispatch: gather selected token rows into bf16 [E*CAP][DM]
// ---------------------------------------------------------------------------
__global__ __launch_bounds__(128) void moe_dispatch(const float* __restrict__ x,
                                                    const int* __restrict__ tok,
                                                    const int* __restrict__ nfill,
                                                    unsigned short* __restrict__ Xd) {
  const int slotId = blockIdx.x;            // e*CAP + slot
  const int e = slotId >> 10;
  const int slot = slotId & (CAP - 1);
  unsigned short* dst = Xd + (size_t)slotId * DM;
  const int d = threadIdx.x * 4;            // 128 threads * 4 = 512 = DM
  if (slot < nfill[e]) {
    const float4 v = *reinterpret_cast<const float4*>(
        x + (size_t)tok[slotId] * DM + d);
    dst[d + 0] = f2bf(v.x); dst[d + 1] = f2bf(v.y);
    dst[d + 2] = f2bf(v.z); dst[d + 3] = f2bf(v.w);
  } else {
    dst[d + 0] = 0; dst[d + 1] = 0; dst[d + 2] = 0; dst[d + 3] = 0;
  }
}

// ---------------------------------------------------------------------------
// GEMM epilogues.  C/D layout: col = nb + (lane&15), row = mb + i + 8*(lane>>4)
// ---------------------------------------------------------------------------
__device__ __forceinline__ void ep_relu_store(v8f c, int e, int mb, int nb,
                                              int lane,
                                              const float* __restrict__ b1,
                                              unsigned short* __restrict__ Hb) {
  const int col = nb + (lane & 15);
  const float bias = b1[e * DFF + col];
  const int rb = mb + ((lane >> 4) << 3);
#pragma unroll
  for (int i = 0; i < 8; ++i) {
    float v = c[i] + bias;
    v = v > 0.f ? v : 0.f;
    Hb[((size_t)e * CAP + rb + i) * DFF + col] = f2bf(v);
  }
}

__device__ __forceinline__ void ep_combine(v8f c, int e, int mb, int nb,
                                           int lane,
                                           const float* __restrict__ b2,
                                           const float* __restrict__ gv,
                                           const int* __restrict__ tok,
                                           float* __restrict__ y) {
  const int col = nb + (lane & 15);
  const float bias = b2[e * DM + col];
  const int rb = mb + ((lane >> 4) << 3);
#pragma unroll
  for (int i = 0; i < 8; ++i) {
    const int r = rb + i;
    const float g = gv[e * CAP + r];      // 0 for unfilled slots -> adds 0
    const int t = tok[e * CAP + r];
    atomicAdd(y + (size_t)t * DM + col, (c[i] + bias) * g);
  }
}

// ---------------------------------------------------------------------------
// 6) FFN layer 1: H = relu(Xd @ W1 + b1)   [E: 1024x512 @ 512x2048]
//    block = 4 waves (2Mx2N), wave tile 64x64 -> block tile 128x128
//    Per K-step: 16 v_wmma, 16 global_load_b128 (4 A-frags, 4 B-frags, 4x reuse)
// ---------------------------------------------------------------------------
__global__ __launch_bounds__(128) void moe_ffn1(const unsigned short* __restrict__ Xd,
                                                const unsigned short* __restrict__ W1bT,
                                                const float* __restrict__ b1,
                                                unsigned short* __restrict__ Hb) {
  const int e = blockIdx.z;
  const int lane = threadIdx.x & 31;
  const int wave = threadIdx.x >> 5;
  const int m0 = blockIdx.y * 128 + (wave >> 1) * 64;
  const int n0 = blockIdx.x * 128 + (wave & 1) * 64;
  const unsigned short* A = Xd + (size_t)e * CAP * DM;
  const unsigned short* B = W1bT + (size_t)e * DFF * DM;
  v8f acc[4][4];
#pragma unroll
  for (int i = 0; i < 4; ++i)
#pragma unroll
    for (int j = 0; j < 4; ++j) acc[i][j] = {};
  for (int k = 0; k < DM; k += 32) {
    if (k + 32 < DM) {  // global_prefetch_b8 of next K-slab
      __builtin_prefetch(A + (size_t)(m0 + (lane & 15)) * DM + k + 32, 0, 1);
      __builtin_prefetch(B + (size_t)(n0 + (lane & 15)) * DM + k + 32, 0, 1);
    }
    v16bf af[4], bfr[4];
#pragma unroll
    for (int i = 0; i < 4; ++i) af[i] = load_a(A, m0 + i * 16, DM, k, lane);
#pragma unroll
    for (int j = 0; j < 4; ++j) bfr[j] = load_b(B, n0 + j * 16, DM, k, lane);
#pragma unroll
    for (int i = 0; i < 4; ++i)
#pragma unroll
      for (int j = 0; j < 4; ++j) acc[i][j] = wmma_bf(af[i], bfr[j], acc[i][j]);
  }
#pragma unroll
  for (int i = 0; i < 4; ++i)
#pragma unroll
    for (int j = 0; j < 4; ++j)
      ep_relu_store(acc[i][j], e, m0 + i * 16, n0 + j * 16, lane, b1, Hb);
}

// ---------------------------------------------------------------------------
// 7) FFN layer 2 + gate-weighted combine: y[tok] += g * (H @ W2 + b2)
//    [E: 1024x2048 @ 2048x512], same 64x64 wave tile
// ---------------------------------------------------------------------------
__global__ __launch_bounds__(128) void moe_ffn2(const unsigned short* __restrict__ Hb,
                                                const unsigned short* __restrict__ W2bT,
                                                const float* __restrict__ b2,
                                                const float* __restrict__ gv,
                                                const int* __restrict__ tok,
                                                float* __restrict__ y) {
  const int e = blockIdx.z;
  const int lane = threadIdx.x & 31;
  const int wave = threadIdx.x >> 5;
  const int m0 = blockIdx.y * 128 + (wave >> 1) * 64;
  const int n0 = blockIdx.x * 128 + (wave & 1) * 64;
  const unsigned short* A = Hb + (size_t)e * CAP * DFF;
  const unsigned short* B = W2bT + (size_t)e * DM * DFF;
  v8f acc[4][4];
#pragma unroll
  for (int i = 0; i < 4; ++i)
#pragma unroll
    for (int j = 0; j < 4; ++j) acc[i][j] = {};
  for (int k = 0; k < DFF; k += 32) {
    if (k + 32 < DFF) {
      __builtin_prefetch(A + (size_t)(m0 + (lane & 15)) * DFF + k + 32, 0, 1);
      __builtin_prefetch(B + (size_t)(n0 + (lane & 15)) * DFF + k + 32, 0, 1);
    }
    v16bf af[4], bfr[4];
#pragma unroll
    for (int i = 0; i < 4; ++i) af[i] = load_a(A, m0 + i * 16, DFF, k, lane);
#pragma unroll
    for (int j = 0; j < 4; ++j) bfr[j] = load_b(B, n0 + j * 16, DFF, k, lane);
#pragma unroll
    for (int i = 0; i < 4; ++i)
#pragma unroll
      for (int j = 0; j < 4; ++j) acc[i][j] = wmma_bf(af[i], bfr[j], acc[i][j]);
  }
#pragma unroll
  for (int i = 0; i < 4; ++i)
#pragma unroll
    for (int j = 0; j < 4; ++j)
      ep_combine(acc[i][j], e, m0 + i * 16, n0 + j * 16, lane, b2, gv, tok, y);
}

// ---------------------------------------------------------------------------
// 8) finalize: y==0 -> EPS; write aux loss
// ---------------------------------------------------------------------------
__global__ __launch_bounds__(256) void moe_finalize(float* __restrict__ out,
                                                    const float* __restrict__ psum,
                                                    const float* __restrict__ cnt1) {
  const int idx = blockIdx.x * 256 + threadIdx.x;
  if (idx < Y_ELEMS) {
    const float v = out[idx];
    out[idx] = (v == 0.0f) ? EPS_F : v;
  }
  if (idx == 0) {
    float acc = 0.f;
    for (int e = 0; e < NE; ++e)
      acc += (cnt1[e] / (float)S_TOK) * (psum[e] / (float)S_TOK);
    out[Y_ELEMS] = 0.1f * (acc / (float)NE);
  }
}

// ---------------------------------------------------------------------------
// Workspace layout (bytes)
// ---------------------------------------------------------------------------
static constexpr size_t OFF_PSUM  = 0;                          // 4 f
static constexpr size_t OFF_CNT1  = 16;                         // 4 f
static constexpr size_t OFF_NFILL = 32;                         // 4 i
static constexpr size_t OFF_TOK   = 64;                         // 4096 i
static constexpr size_t OFF_GV    = OFF_TOK + 4096 * 4;         // 4096 f
static constexpr size_t OFF_E1    = OFF_GV + 4096 * 4;
static constexpr size_t OFF_E2    = OFF_E1 + 4096 * 4;
static constexpr size_t OFF_G1    = OFF_E2 + 4096 * 4;
static constexpr size_t OFF_G2    = OFF_G1 + 4096 * 4;
static constexpr size_t OFF_PROBS = OFF_G2 + 4096 * 4;          // 4096*4 f
static constexpr size_t OFF_XD    = OFF_PROBS + 4096 * 4 * 4;   // bf16
static constexpr size_t OFF_HB    = OFF_XD + (size_t)NE * CAP * DM * 2;
static constexpr size_t OFF_W1T   = OFF_HB + (size_t)NE * CAP * DFF * 2;
static constexpr size_t OFF_W2T   = OFF_W1T + (size_t)NE * DFF * DM * 2;

extern "C" void kernel_launch(void* const* d_in, const int* in_sizes, int n_in,
                              void* d_out, int out_size, void* d_ws, size_t ws_size,
                              hipStream_t stream) {
  const float* x      = (const float*)d_in[0];
  const float* rand_u = (const float*)d_in[1];
  const float* w_gate = (const float*)d_in[2];
  const float* W1     = (const float*)d_in[3];
  const float* b1     = (const float*)d_in[4];
  const float* W2     = (const float*)d_in[5];
  const float* b2     = (const float*)d_in[6];
  float* y = (float*)d_out;

  char* ws = (char*)d_ws;
  float* psum  = (float*)(ws + OFF_PSUM);
  float* cnt1  = (float*)(ws + OFF_CNT1);
  int*   nfill = (int*)  (ws + OFF_NFILL);
  int*   tok   = (int*)  (ws + OFF_TOK);
  float* gv    = (float*)(ws + OFF_GV);
  int*   e1a   = (int*)  (ws + OFF_E1);
  int*   e2a   = (int*)  (ws + OFF_E2);
  float* g1a   = (float*)(ws + OFF_G1);
  float* g2a   = (float*)(ws + OFF_G2);
  float* probs = (float*)(ws + OFF_PROBS);
  unsigned short* Xd   = (unsigned short*)(ws + OFF_XD);
  unsigned short* Hb   = (unsigned short*)(ws + OFF_HB);
  unsigned short* W1bT = (unsigned short*)(ws + OFF_W1T);
  unsigned short* W2bT = (unsigned short*)(ws + OFF_W2T);

  // 1) zero y and slot tables
  moe_init<<<(Y_ELEMS + 255) / 256, 256, 0, stream>>>(y, tok, gv);
  // 2) gating (wave per token)
  moe_gate<<<S_TOK / 4, 128, 0, stream>>>(x, rand_u, w_gate, e1a, e2a, g1a, g2a, probs);
  // 3) deterministic capacity assignment + loss stats
  moe_assign<<<1, 32, 0, stream>>>(e1a, e2a, g1a, g2a, probs, tok, gv, nfill, cnt1, psum);
  // 4) weights fp32 -> bf16 transposed
  moe_cvt_w<<<(NE * DM * DFF) / 256, 256, 0, stream>>>(W1, W2, W1bT, W2bT);
  // 5) gather token rows to bf16
  moe_dispatch<<<NE * CAP, 128, 0, stream>>>(x, tok, nfill, Xd);
  // 6) H = relu(Xd @ W1 + b1)    grid: N/128 x M/128 x E
  moe_ffn1<<<dim3(DFF / 128, CAP / 128, NE), 128, 0, stream>>>(Xd, W1bT, b1, Hb);
  // 7) y[tok] += g * (H @ W2 + b2)
  moe_ffn2<<<dim3(DM / 128, CAP / 128, NE), 128, 0, stream>>>(Hb, W2bT, b2, gv, tok, y);
  // 8) eps fill + loss
  moe_finalize<<<(Y_ELEMS + 255) / 256, 256, 0, stream>>>(y, psum, cnt1);
}
